// SparseAttention_11424613007530
// MI455X (gfx1250) — matrix operand order repaired and checked by reading, not compiled, not run on hardware
//
#include <hip/hip_runtime.h>

#define BB  4
#define TT  8192
#define DD  512
#define HH  8
#define BSZ 16
#define NB  512          // TT / BSZ
#define DH  64           // DD / HH
#define QSCALE 0.044194173824159216f   // DD^-0.5

typedef __attribute__((ext_vector_type(16))) _Float16 v16h;
typedef __attribute__((ext_vector_type(8)))  _Float16 v8h;
typedef __attribute__((ext_vector_type(8)))  float    v8f;
typedef unsigned int u32x4 __attribute__((ext_vector_type(4)));
typedef int          i32x4 __attribute__((ext_vector_type(4)));
typedef int          i32x8 __attribute__((ext_vector_type(8)));

#if defined(__has_builtin)
#if __has_builtin(__builtin_amdgcn_tensor_load_to_lds)
#define HAVE_TDM 1
#endif
#if __has_builtin(__builtin_amdgcn_s_wait_tensorcnt)
#define HAVE_TENSORCNT_BUILTIN 1
#endif
#endif

#define WMMA_F16(A, B, C) \
  __builtin_amdgcn_wmma_f32_16x16x32_f16(false, (A), false, (B), (short)0, (C), false, false)

// Build v16h from two contiguous 8-half chunks (16B loads each).
static __device__ __forceinline__ v16h mk16_pair(const _Float16* p0, const _Float16* p1) {
  v8h lo = *(const v8h*)p0;
  v8h hi = *(const v8h*)p1;
  v16h r;
#pragma unroll
  for (int i = 0; i < 8; ++i) { r[i] = lo[i]; r[8 + i] = hi[i]; }
  return r;
}

// A-layout operand from an fp32 row: element e -> K = kbase + half*16 + e
static __device__ __forceinline__ v16h mkA_f32(const float* row, int kbase, int half) {
  float4 f0 = *(const float4*)(row + kbase + half * 16);
  float4 f1 = *(const float4*)(row + kbase + half * 16 + 4);
  float4 f2 = *(const float4*)(row + kbase + half * 16 + 8);
  float4 f3 = *(const float4*)(row + kbase + half * 16 + 12);
  v16h r;
  r[0]  = (_Float16)f0.x; r[1]  = (_Float16)f0.y; r[2]  = (_Float16)f0.z; r[3]  = (_Float16)f0.w;
  r[4]  = (_Float16)f1.x; r[5]  = (_Float16)f1.y; r[6]  = (_Float16)f1.z; r[7]  = (_Float16)f1.w;
  r[8]  = (_Float16)f2.x; r[9]  = (_Float16)f2.y; r[10] = (_Float16)f2.z; r[11] = (_Float16)f2.w;
  r[12] = (_Float16)f3.x; r[13] = (_Float16)f3.y; r[14] = (_Float16)f3.z; r[15] = (_Float16)f3.w;
  return r;
}

// B-layout operand from 16 contiguous fp32: element e -> K = base + half*16 + e (half via pointer)
static __device__ __forceinline__ v16h mkB_f32(const float* p) {
  float4 g0 = *(const float4*)(p);
  float4 g1 = *(const float4*)(p + 4);
  float4 g2 = *(const float4*)(p + 8);
  float4 g3 = *(const float4*)(p + 12);
  v16h r;
  r[0]  = (_Float16)g0.x; r[1]  = (_Float16)g0.y; r[2]  = (_Float16)g0.z; r[3]  = (_Float16)g0.w;
  r[4]  = (_Float16)g1.x; r[5]  = (_Float16)g1.y; r[6]  = (_Float16)g1.z; r[7]  = (_Float16)g1.w;
  r[8]  = (_Float16)g2.x; r[9]  = (_Float16)g2.y; r[10] = (_Float16)g2.z; r[11] = (_Float16)g2.w;
  r[12] = (_Float16)g3.x; r[13] = (_Float16)g3.y; r[14] = (_Float16)g3.z; r[15] = (_Float16)g3.w;
  return r;
}

#ifdef HAVE_TDM
// Issue one TDM descriptor: contiguous `bytes` from global -> LDS offset.
// 1-D tile of 8-byte elements: tile_dim0 = tensor_dim0 = bytes/8.
static __device__ __forceinline__ void tdm_load_contig(const void* gsrc,
                                                       unsigned lds_off,
                                                       unsigned bytes) {
  unsigned long long ga = (unsigned long long)(uintptr_t)gsrc;
  unsigned n8 = bytes >> 3;  // # of 8-byte elements (8192 for 64KB)
  u32x4 g0 = {};
  g0[0] = 1u;                                     // count=1, user mode, no gather
  g0[1] = lds_off;                                // lds_addr (bytes)
  g0[2] = (unsigned)(ga & 0xffffffffu);           // global_addr[31:0]
  g0[3] = (unsigned)((ga >> 32) & 0x01ffffffu)    // global_addr[56:32]
          | (2u << 30);                           // type = 2 ("image")
  i32x8 g1 = {};
  g1[0] = (3 << 16);                              // wg_mask=0, data_size=3 (8B)
  g1[1] = (int)((n8 & 0xffffu) << 16);            // tensor_dim0[15:0]  @ bits 63:48
  g1[2] = (int)((n8 >> 16) & 0xffffu);            // tensor_dim0[31:16] @ bits 79:64
  g1[3] = (int)((n8 & 0xffffu) << 16);            // tile_dim0 @ bits 127:112
  g1[5] = (int)n8;                                // tensor_dim0_stride[31:0]
  i32x4 g2 = {};
  i32x4 g3 = {};
#if defined(__clang_major__) && (__clang_major__ >= 23)
  i32x8 g4 = {};
  __builtin_amdgcn_tensor_load_to_lds(g0, g1, g2, g3, g4, 0);
#else
  __builtin_amdgcn_tensor_load_to_lds(g0, g1, g2, g3, 0);
#endif
}
#endif

static __device__ __forceinline__ void wait_tensorcnt0() {
#ifdef HAVE_TENSORCNT_BUILTIN
  __builtin_amdgcn_s_wait_tensorcnt(0);
#else
  asm volatile("s_wait_tensorcnt 0x0" ::: "memory");
#endif
}

// ---------------------------------------------------------------------------
// Kernel 1: fused QKV projection.
// Grid: BB*BSZ*(NB/16) = 2048 workgroups of 256 threads (8 waves).
// Workgroup (b, s, ngrp): rows n = ngrp*16 .. +15, x row t = n*BSZ + s.
// Wave w owns 12 of 96 column tiles (q:0-31, k:32-63, v:64-95).
// Q/K stored f16 as [combo][NB][DH]; V stored transposed [combo][DH][NB].
// ---------------------------------------------------------------------------
__global__ __launch_bounds__(256) void qkv_proj_kernel(
    const float* __restrict__ x,
    const float* __restrict__ Wq, const float* __restrict__ bq,
    const float* __restrict__ Wk, const float* __restrict__ bk,
    const float* __restrict__ Wv, const float* __restrict__ bv,
    _Float16* __restrict__ qws, _Float16* __restrict__ kws,
    _Float16* __restrict__ vtws) {
  extern __shared__ __align__(16) char smem_raw[];
  float* alds = (float*)smem_raw;  // [16][512] fp32 x-tile (32 KB)

  const int wg    = blockIdx.x;
  const int ngrp  = wg & 31;
  const int s     = (wg >> 5) & 15;
  const int b     = wg >> 9;
  const int nbase = ngrp * 16;
  const int tid   = threadIdx.x;

  // Cooperative stage of the 16x512 x tile (rows strided by BSZ*DD in memory).
#pragma unroll
  for (int i = 0; i < 8; ++i) {
    int flat = i * 256 + tid;          // float4 index, 16 rows * 128 each
    int row  = flat >> 7;
    int c4   = flat & 127;
    float4 v = *(const float4*)(x + (size_t)b * TT * DD +
                                (size_t)((nbase + row) * BSZ + s) * DD + c4 * 4);
    ((float4*)alds)[row * 128 + c4] = v;
  }
  __syncthreads();

  const int wave = tid >> 5;
  const int lane = tid & 31;
  const int half = lane >> 4;
  const int m16  = lane & 15;

  v8f acc[12];
#pragma unroll
  for (int t = 0; t < 12; ++t) acc[t] = v8f{};

  const float* wrow[12];
#pragma unroll
  for (int t = 0; t < 12; ++t) {
    int tt = wave * 12 + t;
    int mat = tt >> 5;
    int colbase = (tt & 31) * 16;
    const float* W = (mat == 0) ? Wq : ((mat == 1) ? Wk : Wv);
    wrow[t] = W + (size_t)(colbase + m16) * DD;   // B[k][n] = W[n][k]
  }

  const float* arow = alds + m16 * DD;
  for (int ks = 0; ks < 16; ++ks) {
    int kbase = ks * 32;
    v16h a = mkA_f32(arow, kbase, half);
#pragma unroll
    for (int t = 0; t < 12; ++t) {
      v16h bm = mkB_f32(wrow[t] + kbase + half * 16);
      acc[t] = WMMA_F16(a, bm, acc[t]);
    }
  }

  // Epilogue: bias, fold logit scale into Q, write f16 in attention layouts.
#pragma unroll
  for (int t = 0; t < 12; ++t) {
    int tt = wave * 12 + t;
    int mat = tt >> 5;
    int colbase = (tt & 31) * 16;
    int col = colbase + m16;
    const float* bias = (mat == 0) ? bq : ((mat == 1) ? bk : bv);
    float bvv = bias[col];
    float sc  = (mat == 0) ? QSCALE : 1.0f;
    int h  = col >> 6;
    int dd = col & 63;
    size_t combo = ((size_t)((b * BSZ + s) * HH + h)) * (NB * DH);
#pragma unroll
    for (int r = 0; r < 8; ++r) {
      int n = nbase + r + half * 8;           // C-layout: row M = r + 8*half
      _Float16 hv = (_Float16)((acc[t][r] + bvv) * sc);
      if (mat == 2)      vtws[combo + (size_t)dd * NB + n] = hv;
      else if (mat == 1) kws[combo + (size_t)n * DH + dd]  = hv;
      else               qws[combo + (size_t)n * DH + dd]  = hv;
    }
  }
}

// ---------------------------------------------------------------------------
// Kernel 2: per-(b,s,h) attention over 512 blocks, flash-style two passes.
// Grid: 512 combos * 4 rowgroups = 2048 workgroups of 256 threads (8 waves).
// LDS: K f16 [512][64] (64KB) + V^T f16 [64][512] (64KB) + P stage 8*[16][32].
// K/V^T staged by the Tensor Data Mover (one descriptor each, TENSORcnt).
// ---------------------------------------------------------------------------
__global__ __launch_bounds__(256) void blk_attn_kernel(
    const _Float16* __restrict__ qws, const _Float16* __restrict__ kws,
    const _Float16* __restrict__ vtws, float* __restrict__ out) {
  extern __shared__ __align__(16) char smem_raw[];
  _Float16* Kl  = (_Float16*)smem_raw;      // [512][64]
  _Float16* Vl  = Kl + NB * DH;             // [64][512] (V transposed)
  _Float16* Pst = Vl + NB * DH;             // [8][16][32]

  const int wg    = blockIdx.x;
  const int rg    = wg & 3;
  const int combo = wg >> 2;                // ((b*16+s)*8 + h)
  const int h     = combo & 7;
  const int s     = (combo >> 3) & 15;
  const int b     = combo >> 7;
  const size_t base = (size_t)combo * (NB * DH);
  const int tid = threadIdx.x;

#ifdef HAVE_TDM
  // Tensor Data Mover: wave 0 issues both 64KB contiguous tile loads.
  if (tid < 32) {
    tdm_load_contig(kws + base,  (unsigned)(uintptr_t)Kl, NB * DH * 2);
    tdm_load_contig(vtws + base, (unsigned)(uintptr_t)Vl, NB * DH * 2);
    wait_tensorcnt0();
  }
  __syncthreads();
#else
  // Fallback: VGPR-staged cooperative copy.
  {
    const uint4* ksrc = (const uint4*)(kws + base);
    const uint4* vsrc = (const uint4*)(vtws + base);
    uint4* kdst = (uint4*)Kl;
    uint4* vdst = (uint4*)Vl;
#pragma unroll
    for (int i = 0; i < 8; ++i) {
      kdst[i * 256 + tid] = ksrc[i * 256 + tid];
      vdst[i * 256 + tid] = vsrc[i * 256 + tid];
    }
    __syncthreads();
  }
#endif

  const int wave = tid >> 5;
  const int lane = tid & 31;
  const int half = lane >> 4;
  const int m16  = lane & 15;
  const int rowbase = rg * 128 + wave * 16;

  // Q tile (16 rows x 64) in A-layout registers; logit scale already folded.
  const _Float16* qrow = qws + base + (size_t)(rowbase + m16) * DH;
  v16h qa0 = mk16_pair(qrow + half * 16,      qrow + half * 16 + 8);
  v16h qa1 = mk16_pair(qrow + 32 + half * 16, qrow + 32 + half * 16 + 8);

  float mrun[8], lrun[8];
#pragma unroll
  for (int r = 0; r < 8; ++r) { mrun[r] = -3.0e38f; lrun[r] = 0.0f; }

  // ---- Pass 1: row max + sum of exp (online) ----
  for (int j = 0; j < 32; ++j) {
    const _Float16* krow = Kl + (j * 16 + m16) * DH;   // B lane: n = j*16 + m16
    v16h b0 = mk16_pair(krow + half * 16,      krow + half * 16 + 8);
    v16h b1 = mk16_pair(krow + 32 + half * 16, krow + 32 + half * 16 + 8);
    v8f c = v8f{};
    c = WMMA_F16(qa0, b0, c);
    c = WMMA_F16(qa1, b1, c);
#pragma unroll
    for (int r = 0; r < 8; ++r) {
      float v = c[r];
      float tm = v;
      tm = fmaxf(tm, __shfl_xor(tm, 1));
      tm = fmaxf(tm, __shfl_xor(tm, 2));
      tm = fmaxf(tm, __shfl_xor(tm, 4));
      tm = fmaxf(tm, __shfl_xor(tm, 8));
      float nm = fmaxf(mrun[r], tm);
      float p  = __expf(v - nm);
      p += __shfl_xor(p, 1);
      p += __shfl_xor(p, 2);
      p += __shfl_xor(p, 4);
      p += __shfl_xor(p, 8);
      lrun[r] = lrun[r] * __expf(mrun[r] - nm) + p;
      mrun[r] = nm;
    }
  }

  // ---- Pass 2: recompute S, P = exp(S - m), O += P @ V ----
  v8f oacc[4];
#pragma unroll
  for (int dt = 0; dt < 4; ++dt) oacc[dt] = v8f{};
  _Float16* pst = Pst + wave * (16 * 32);

  for (int cchunk = 0; cchunk < 16; ++cchunk) {
#pragma unroll
    for (int jj = 0; jj < 2; ++jj) {
      int j = cchunk * 2 + jj;
      const _Float16* krow = Kl + (j * 16 + m16) * DH;
      v16h b0 = mk16_pair(krow + half * 16,      krow + half * 16 + 8);
      v16h b1 = mk16_pair(krow + 32 + half * 16, krow + 32 + half * 16 + 8);
      v8f c = v8f{};
      c = WMMA_F16(qa0, b0, c);
      c = WMMA_F16(qa1, b1, c);
#pragma unroll
      for (int r = 0; r < 8; ++r) {
        float p = __expf(c[r] - mrun[r]);
        pst[(r + half * 8) * 32 + jj * 16 + m16] = (_Float16)p;  // C-layout scatter
      }
    }
    asm volatile("s_wait_dscnt 0x0" ::: "memory");   // stores -> A-layout reload
    v16h pa = mk16_pair(pst + m16 * 32 + half * 16,
                        pst + m16 * 32 + half * 16 + 8);
#pragma unroll
    for (int dt = 0; dt < 4; ++dt) {
      const _Float16* vrow = Vl + (dt * 16 + m16) * NB + cchunk * 32 + half * 16;
      v16h vb = mk16_pair(vrow, vrow + 8);
      oacc[dt] = WMMA_F16(pa, vb, oacc[dt]);
    }
    asm volatile("s_wait_dscnt 0x0" ::: "memory");   // reads done before next overwrite
  }

  // ---- Epilogue: divide by softmax denom, scatter to out[b, n*16+s, h*64+d] ----
#pragma unroll
  for (int dt = 0; dt < 4; ++dt) {
#pragma unroll
    for (int r = 0; r < 8; ++r) {
      int n = rowbase + r + half * 8;
      int t = n * BSZ + s;
      int d = h * DH + dt * 16 + m16;
      out[(size_t)b * TT * DD + (size_t)t * DD + d] = oacc[dt][r] / lrun[r];
    }
  }
}

// ---------------------------------------------------------------------------
extern "C" void kernel_launch(void* const* d_in, const int* in_sizes, int n_in,
                              void* d_out, int out_size, void* d_ws, size_t ws_size,
                              hipStream_t stream) {
  const float* x  = (const float*)d_in[0];
  const float* Wq = (const float*)d_in[1];
  const float* bq = (const float*)d_in[2];
  const float* Wk = (const float*)d_in[3];
  const float* bk = (const float*)d_in[4];
  const float* Wv = (const float*)d_in[5];
  const float* bv = (const float*)d_in[6];
  float* out = (float*)d_out;

  const size_t elems = (size_t)BB * BSZ * HH * NB * DH;  // 16M f16 per array
  _Float16* qws  = (_Float16*)d_ws;
  _Float16* kws  = qws + elems;
  _Float16* vtws = kws + elems;

  dim3 blk(256);
  dim3 grid_proj(BB * BSZ * (NB / 16));          // 2048
  size_t lds_proj = 16 * DD * sizeof(float);     // 32 KB
  qkv_proj_kernel<<<grid_proj, blk, lds_proj, stream>>>(
      x, Wq, bq, Wk, bk, Wv, bv, qws, kws, vtws);

  dim3 grid_attn(BB * BSZ * HH * 4);             // 2048
  size_t lds_attn = (size_t)(NB * DH * 2 + 8 * 16 * 32) * sizeof(_Float16);  // 136 KB
  blk_attn_kernel<<<grid_attn, blk, lds_attn, stream>>>(qws, kws, vtws, out);
}
